// Full_Tmix_Params_35270271435415
// MI455X (gfx1250) — compile-verified, matrix operand order checked by reading, not run on hardware
//
#include <hip/hip_runtime.h>
#include <hip/hip_bf16.h>
#include <math.h>

// ---------------- problem constants ----------------
#define BB   2
#define TT   4096
#define CCH  2048
#define HSZ  64
#define DWH  128
#define DAH  128
#define DVH  64
#define DGH  224
#define MROWS (BB*TT)                       // 8192 token rows

typedef __attribute__((ext_vector_type(16))) __bf16 v16bf;
typedef __attribute__((ext_vector_type(8)))  __bf16 v8bf;
typedef __attribute__((ext_vector_type(8)))  float  v8f;

static __device__ __forceinline__ unsigned short f2bf(float f) {
  union { float f; unsigned u; } v; v.f = f;
  unsigned r = v.u + 0x7FFFu + ((v.u >> 16) & 1u);   // round-to-nearest-even
  return (unsigned short)(r >> 16);
}
static __device__ __forceinline__ float sigmoidf_(float z) {
  return 1.f / (1.f + expf(-z));
}
static __device__ __forceinline__ unsigned pack2(float a, float b) {
  return (unsigned)f2bf(a) | ((unsigned)f2bf(b) << 16);
}

// ---------------- generic WMMA GEMM ----------------
// out[M,N] = A[M,K] * B[N,K]^T  (B row-major [N,K] bf16, contiguous K)
// AMODE 0: A = fp32 x with fused time-shift mix (coeff[K]) -> bf16
// AMODE 1: A = bf16 row-major [M,K]
// EPI  0: f32 store            1: bf16 store
//      2: tanh -> bf16         3: sigmoid -> bf16
//      4: w = -softplus(-(w0+acc)) - 0.5  (f32, bias)
//      5: sigmoid(bias + acc)  (f32)
//      6: in-place v mix: C = C + (vfirst - C) * sigmoid(bias + acc)
#define BM 128
#define BN 128
#define BK 32
#define LDS_STRIDE 40   // bf16 elements; 80B rows keep 16B alignment, skew banks

template<int AMODE, int EPI>
__global__ __launch_bounds__(256)
void gemm_wmma(const void* __restrict__ Aptr, const float* __restrict__ coeff,
               const unsigned short* __restrict__ Bmat,
               void* __restrict__ Cout, const float* __restrict__ bias,
               const float* __restrict__ vfirst,
               int M, int N, int K)
{
  __shared__ unsigned short sA[BM * LDS_STRIDE];
  __shared__ unsigned short sB[BN * LDS_STRIDE];

  const int tid  = threadIdx.x;
  const int lane = tid & 31;
  const int wave = tid >> 5;          // 8 waves
  const int wm   = wave >> 1;         // 0..3 (M direction)
  const int wn   = wave & 1;          // 0..1 (N direction)

  const int ar = tid >> 1;            // 0..127 tile row loaded by this thread
  const int ac = (tid & 1) * 16;      // 0 or 16 col base (16 bf16 per thread)

  const int m0 = blockIdx.y * BM;
  const int n0 = blockIdx.x * BN;

  v8f acc[2][4];
  #pragma unroll
  for (int i = 0; i < 2; i++)
    #pragma unroll
    for (int j = 0; j < 4; j++)
      #pragma unroll
      for (int e = 0; e < 8; e++) acc[i][j][e] = 0.f;

  const int grow = m0 + ar;                       // global A row (< M)
  const bool has_prev = (grow % TT) != 0;         // time-shift boundary
  const float pscale = has_prev ? 1.f : 0.f;      // neutralize prev at t==0
  const int prev_off = has_prev ? K : 0;          // safe clamped offset
  const int gbn = n0 + ar;                        // global B row (output col)
  const bool bvalid = (gbn < N);
  const int gbn_c = bvalid ? gbn : (N - 1);       // clamped, always in range

  for (int kc = 0; kc < K; kc += BK) {
    __syncthreads();
    // ---------------- fill A tile ----------------
    if (AMODE == 0) {
      const float* xb = (const float*)Aptr + (size_t)grow * K + kc + ac;
      const float4* xv = (const float4*)xb;
      const float4* pv = (const float4*)(xb - prev_off);   // == xb at t==0
      const float4* cv = (const float4*)(coeff + kc + ac);
      unsigned pk[8];
      #pragma unroll
      for (int q = 0; q < 4; q++) {
        float4 xq = xv[q];
        float4 pq = pv[q];
        float4 cq = cv[q];
        float p0 = pq.x * pscale, p1 = pq.y * pscale;
        float p2 = pq.z * pscale, p3 = pq.w * pscale;
        pk[2*q + 0] = pack2(xq.x + (p0 - xq.x) * cq.x,
                            xq.y + (p1 - xq.y) * cq.y);
        pk[2*q + 1] = pack2(xq.z + (p2 - xq.z) * cq.z,
                            xq.w + (p3 - xq.w) * cq.w);
      }
      *(uint4*)&sA[ar * LDS_STRIDE + ac]     = make_uint4(pk[0], pk[1], pk[2], pk[3]);
      *(uint4*)&sA[ar * LDS_STRIDE + ac + 8] = make_uint4(pk[4], pk[5], pk[6], pk[7]);
      if (kc + BK < K) __builtin_prefetch((const void*)(xb + BK), 0, 1);
    } else {
      const unsigned short* A = (const unsigned short*)Aptr;
      const uint4* ap = (const uint4*)(A + (size_t)grow * K + kc + ac);
      *(uint4*)&sA[ar * LDS_STRIDE + ac]     = ap[0];
      *(uint4*)&sA[ar * LDS_STRIDE + ac + 8] = ap[1];
    }
    // ---------------- fill B tile (clamped row, branch-free zero) ----------------
    {
      const uint4* bp = (const uint4*)(Bmat + (size_t)gbn_c * K + kc + ac);
      uint4 b0 = bp[0];
      uint4 b1 = bp[1];
      if (!bvalid) {                       // register zeroing only, no loads
        b0 = make_uint4(0u, 0u, 0u, 0u);
        b1 = make_uint4(0u, 0u, 0u, 0u);
      }
      *(uint4*)&sB[ar * LDS_STRIDE + ac]     = b0;
      *(uint4*)&sB[ar * LDS_STRIDE + ac + 8] = b1;
      if (kc + BK < K)
        __builtin_prefetch((const void*)(Bmat + (size_t)gbn_c * K + kc + BK + ac), 0, 1);
    }
    __syncthreads();

    // ---------------- compute 2x4 WMMA tiles ----------------
    const int rsel = lane & 15;
    const int kb   = ((lane >> 4) & 1) * 8;   // lanes 0-15: K 0-7/16-23; 16-31: K 8-15/24-31

    v16bf fa[2], fb[4];
    #pragma unroll
    for (int i = 0; i < 2; i++) {
      const unsigned short* p = &sA[(wm * 32 + i * 16 + rsel) * LDS_STRIDE + kb];
      v8bf lo = *(const v8bf*)p;
      v8bf hi = *(const v8bf*)(p + 16);
      fa[i] = __builtin_shufflevector(lo, hi, 0,1,2,3,4,5,6,7,8,9,10,11,12,13,14,15);
    }
    #pragma unroll
    for (int j = 0; j < 4; j++) {
      const unsigned short* p = &sB[(wn * 64 + j * 16 + rsel) * LDS_STRIDE + kb];
      v8bf lo = *(const v8bf*)p;
      v8bf hi = *(const v8bf*)(p + 16);
      fb[j] = __builtin_shufflevector(lo, hi, 0,1,2,3,4,5,6,7,8,9,10,11,12,13,14,15);
    }
    #pragma unroll
    for (int i = 0; i < 2; i++)
      #pragma unroll
      for (int j = 0; j < 4; j++)
        acc[i][j] = __builtin_amdgcn_wmma_f32_16x16x32_bf16(
            false, fa[i], false, fb[j], (short)0, acc[i][j], false, false);
  }

  // ---------------- epilogue ----------------
  const int tmb = m0 + wm * 32;
  const int tnb = n0 + wn * 64;
  #pragma unroll
  for (int i = 0; i < 2; i++) {
    #pragma unroll
    for (int j = 0; j < 4; j++) {
      #pragma unroll
      for (int r = 0; r < 8; r++) {
        int row = tmb + i * 16 + ((lane < 16) ? r : r + 8);
        int col = tnb + j * 16 + (lane & 15);
        if (col < N) {
          size_t idx = (size_t)row * N + col;
          float a = acc[i][j][r];
          if constexpr (EPI == 0) {
            ((float*)Cout)[idx] = a;
          } else if constexpr (EPI == 1) {
            ((unsigned short*)Cout)[idx] = f2bf(a);
          } else if constexpr (EPI == 2) {
            ((unsigned short*)Cout)[idx] = f2bf(tanhf(a));
          } else if constexpr (EPI == 3) {
            ((unsigned short*)Cout)[idx] = f2bf(sigmoidf_(a));
          } else if constexpr (EPI == 4) {
            float z = bias[col] + a;
            ((float*)Cout)[idx] = -log1pf(expf(-z)) - 0.5f;
          } else if constexpr (EPI == 5) {
            ((float*)Cout)[idx] = sigmoidf_(bias[col] + a);
          } else if constexpr (EPI == 6) {
            float gate = sigmoidf_(bias[col] + a);
            float vr = ((float*)Cout)[idx];          // v_raw written earlier
            float vf = vfirst[idx];
            ((float*)Cout)[idx] = vr + (vf - vr) * gate;
          }
        }
      }
    }
  }
}

// ---------------- helper kernels ----------------
__global__ void cvt_bf16(const float* __restrict__ in, unsigned short* __restrict__ out, size_t n) {
  size_t i = (size_t)blockIdx.x * blockDim.x + threadIdx.x;
  size_t stride = (size_t)gridDim.x * blockDim.x;
  for (; i < n; i += stride) out[i] = f2bf(in[i]);
}

__global__ void transpose_cvt(const float* __restrict__ in, unsigned short* __restrict__ out,
                              int R, int Ccols) {
  int idx = blockIdx.x * blockDim.x + threadIdx.x;
  if (idx < R * Ccols) {
    int r = idx / Ccols, c = idx % Ccols;
    out[(size_t)c * R + r] = f2bf(in[idx]);
  }
}

__global__ void copy4(const float4* __restrict__ in, float4* __restrict__ out, size_t n4) {
  size_t i = (size_t)blockIdx.x * blockDim.x + threadIdx.x;
  if (i < n4) out[i] = in[i];
}

// in-place: kio holds k_raw -> becomes k; kkout gets per-head-normalized k*k_k
__global__ __launch_bounds__(256)
void final_k(float* __restrict__ kio, float* __restrict__ kkout,
             const float* __restrict__ ain,
             const float* __restrict__ k_k, const float* __restrict__ k_a) {
  __shared__ float sm[256];
  size_t idx = (size_t)blockIdx.x * 256 + threadIdx.x;
  int c = (int)(idx % CCH);
  float kraw = kio[idx];
  float kkv  = kraw * k_k[c];
  sm[threadIdx.x] = kkv * kkv;
  __syncthreads();
  int base = threadIdx.x & ~63;           // 64-wide head groups align with blocks
  float s = 0.f;
  #pragma unroll
  for (int i = 0; i < 64; i++) s += sm[base + i];
  float denom = fmaxf(sqrtf(s), 1e-12f);
  kkout[idx] = kkv / denom;
  float a = ain[idx];
  kio[idx] = kraw * (1.f + (a - 1.f) * k_a[c]);
}

// ---------------- host-side launch ----------------
extern "C" void kernel_launch(void* const* d_in, const int* in_sizes, int n_in,
                              void* d_out, int out_size, void* d_ws, size_t ws_size,
                              hipStream_t stream) {
  (void)in_sizes; (void)n_in; (void)out_size; (void)ws_size;
  const float* x   = (const float*)d_in[0];
  const float* vfi = (const float*)d_in[1];
  const float* x_r = (const float*)d_in[2];
  const float* x_w = (const float*)d_in[3];
  const float* x_k = (const float*)d_in[4];
  const float* x_v = (const float*)d_in[5];
  const float* x_a = (const float*)d_in[6];
  const float* x_g = (const float*)d_in[7];
  const float* w0  = (const float*)d_in[8];
  const float* w1  = (const float*)d_in[9];
  const float* w2  = (const float*)d_in[10];
  const float* a0  = (const float*)d_in[11];
  const float* a1  = (const float*)d_in[12];
  const float* a2  = (const float*)d_in[13];
  const float* v0  = (const float*)d_in[14];
  const float* v1  = (const float*)d_in[15];
  const float* v2  = (const float*)d_in[16];
  const float* g1  = (const float*)d_in[17];
  const float* g2  = (const float*)d_in[18];
  const float* kkc = (const float*)d_in[19];
  const float* kac = (const float*)d_in[20];
  const float* Wr  = (const float*)d_in[21];
  const float* Wk  = (const float*)d_in[22];
  const float* Wv  = (const float*)d_in[23];

  const size_t S = (size_t)MROWS * CCH;   // elements per output tensor
  float* out  = (float*)d_out;
  float* o_r  = out + 0 * S;
  float* o_w  = out + 1 * S;
  float* o_k  = out + 2 * S;
  float* o_v  = out + 3 * S;
  float* o_a  = out + 4 * S;
  float* o_g  = out + 5 * S;
  float* o_kk = out + 6 * S;
  float* o_vf = out + 7 * S;

  // workspace carve-out (~37 MB total)
  unsigned char* ws = (unsigned char*)d_ws;
  size_t off = 0;
  auto carve = [&](size_t bytes) -> void* {
    void* p = ws + off; off += (bytes + 255) & ~(size_t)255; return p;
  };
  unsigned short* wsWr = (unsigned short*)carve((size_t)CCH * CCH * 2);
  unsigned short* wsWk = (unsigned short*)carve((size_t)CCH * CCH * 2);
  unsigned short* wsWv = (unsigned short*)carve((size_t)CCH * CCH * 2);
  unsigned short* w1t  = (unsigned short*)carve((size_t)DWH * CCH * 2);
  unsigned short* a1t  = (unsigned short*)carve((size_t)DAH * CCH * 2);
  unsigned short* v1t  = (unsigned short*)carve((size_t)DVH * CCH * 2);
  unsigned short* g1t  = (unsigned short*)carve((size_t)DGH * CCH * 2);
  unsigned short* w2t  = (unsigned short*)carve((size_t)CCH * DWH * 2);
  unsigned short* a2t  = (unsigned short*)carve((size_t)CCH * DAH * 2);
  unsigned short* v2t  = (unsigned short*)carve((size_t)CCH * DVH * 2);
  unsigned short* g2t  = (unsigned short*)carve((size_t)CCH * DGH * 2);
  unsigned short* hw   = (unsigned short*)carve((size_t)MROWS * DWH * 2);
  unsigned short* ha   = (unsigned short*)carve((size_t)MROWS * DAH * 2);
  unsigned short* hv   = (unsigned short*)carve((size_t)MROWS * DVH * 2);
  unsigned short* hg   = (unsigned short*)carve((size_t)MROWS * DGH * 2);

  // ---- phase 0: weight conversions ----
  cvt_bf16<<<4096, 256, 0, stream>>>(Wr, wsWr, (size_t)CCH * CCH);
  cvt_bf16<<<4096, 256, 0, stream>>>(Wk, wsWk, (size_t)CCH * CCH);
  cvt_bf16<<<4096, 256, 0, stream>>>(Wv, wsWv, (size_t)CCH * CCH);
  transpose_cvt<<<(CCH*DWH + 255)/256, 256, 0, stream>>>(w1, w1t, CCH, DWH);
  transpose_cvt<<<(CCH*DAH + 255)/256, 256, 0, stream>>>(a1, a1t, CCH, DAH);
  transpose_cvt<<<(CCH*DVH + 255)/256, 256, 0, stream>>>(v1, v1t, CCH, DVH);
  transpose_cvt<<<(CCH*DGH + 255)/256, 256, 0, stream>>>(g1, g1t, CCH, DGH);
  transpose_cvt<<<(DWH*CCH + 255)/256, 256, 0, stream>>>(w2, w2t, DWH, CCH);
  transpose_cvt<<<(DAH*CCH + 255)/256, 256, 0, stream>>>(a2, a2t, DAH, CCH);
  transpose_cvt<<<(DVH*CCH + 255)/256, 256, 0, stream>>>(v2, v2t, DVH, CCH);
  transpose_cvt<<<(DGH*CCH + 255)/256, 256, 0, stream>>>(g2, g2t, DGH, CCH);

  const dim3 blk(256);
  const dim3 gBig((CCH + BN - 1) / BN, MROWS / BM);       // 16 x 64
  const dim3 gN128(1, MROWS / BM);
  const dim3 gN64(1, MROWS / BM);
  const dim3 gN224((DGH + BN - 1) / BN, MROWS / BM);      // 2 x 64

  // ---- phase 1: big projections + LoRA hidden layers (A = fused mix of x) ----
  gemm_wmma<0,0><<<gBig, blk, 0, stream>>>(x, x_r, wsWr, o_r, nullptr, nullptr, MROWS, CCH, CCH);
  gemm_wmma<0,0><<<gBig, blk, 0, stream>>>(x, x_k, wsWk, o_k, nullptr, nullptr, MROWS, CCH, CCH);
  gemm_wmma<0,0><<<gBig, blk, 0, stream>>>(x, x_v, wsWv, o_v, nullptr, nullptr, MROWS, CCH, CCH);
  gemm_wmma<0,2><<<gN128, blk, 0, stream>>>(x, x_w, w1t, hw, nullptr, nullptr, MROWS, DWH, CCH);
  gemm_wmma<0,1><<<gN128, blk, 0, stream>>>(x, x_a, a1t, ha, nullptr, nullptr, MROWS, DAH, CCH);
  gemm_wmma<0,1><<<gN64,  blk, 0, stream>>>(x, x_v, v1t, hv, nullptr, nullptr, MROWS, DVH, CCH);
  gemm_wmma<0,3><<<gN224, blk, 0, stream>>>(x, x_g, g1t, hg, nullptr, nullptr, MROWS, DGH, CCH);

  // ---- phase 2: LoRA output layers with fused epilogues ----
  gemm_wmma<1,4><<<gBig, blk, 0, stream>>>(hw, nullptr, w2t, o_w, w0, nullptr, MROWS, CCH, DWH);
  gemm_wmma<1,5><<<gBig, blk, 0, stream>>>(ha, nullptr, a2t, o_a, a0, nullptr, MROWS, CCH, DAH);
  gemm_wmma<1,6><<<gBig, blk, 0, stream>>>(hv, nullptr, v2t, o_v, v0, vfi,     MROWS, CCH, DVH);
  gemm_wmma<1,0><<<gBig, blk, 0, stream>>>(hg, nullptr, g2t, o_g, nullptr, nullptr, MROWS, CCH, DGH);

  // ---- phase 3: k / kk finalize + v_first passthrough ----
  final_k<<<(unsigned)(S / 256), 256, 0, stream>>>(o_k, o_kk, o_a, kkc, kac);
  copy4<<<(unsigned)(S / 4 / 256), 256, 0, stream>>>((const float4*)vfi, (float4*)o_vf, S / 4);
}
